// OutputDeliberationAndGenerationControl_25039659335787
// MI455X (gfx1250) — compile-verified, matrix operand order checked by reading, not run on hardware
//
#include <hip/hip_runtime.h>
#include <math.h>

// Problem constants (B=2,S=1024,H=512,V=50257)
#define VOCAB 50257
#define HDIM  512
#define MROWS 2048            // B*S
#define INVT  (1.0f/0.7f)
#define TOPK  50
#define TOPP  0.9f

typedef __attribute__((ext_vector_type(16))) __bf16        v16bf;
typedef __attribute__((ext_vector_type(8)))  float         v8f;
typedef __attribute__((ext_vector_type(4)))  unsigned int  u32x4;
typedef __attribute__((ext_vector_type(8)))  int           i32x8;
typedef __attribute__((ext_vector_type(4)))  int           i32x4;

// ---------------------------------------------------------------------------
// fp32 -> bf16 (round-to-nearest-even)
// ---------------------------------------------------------------------------
__device__ __forceinline__ unsigned short f2bf(float f) {
  unsigned u = __float_as_uint(f);
  unsigned r = u + 0x7FFFu + ((u >> 16) & 1u);
  return (unsigned short)(r >> 16);
}

__global__ __launch_bounds__(256) void cvt_f32_to_bf16_kernel(
    const float* __restrict__ in, unsigned short* __restrict__ out, int n4) {
  int i = blockIdx.x * 256 + threadIdx.x;
  if (i < n4) {
    float4 v = ((const float4*)in)[i];
    unsigned lo = (unsigned)f2bf(v.x) | ((unsigned)f2bf(v.y) << 16);
    unsigned hi = (unsigned)f2bf(v.z) | ((unsigned)f2bf(v.w) << 16);
    uint2 p; p.x = lo; p.y = hi;
    ((uint2*)out)[i] = p;
  }
}

// ---------------------------------------------------------------------------
// bf16 WMMA GEMM: out[m, v] = (x[m,:] . W[v,:] + b[v]) / T
// block tile 128(M) x 128(N), K-step 32, double-buffered LDS.
// 8 waves in a 2(M) x 4(N) grid; each wave owns 64M x 32N:
//   4 A-frags + 2 B-frags (12 ds_load_b128) -> 8 WMMAs per K-step.
// LDS row stride = 40 elements (80 B) -> conflict-free b128 lane access.
// ---------------------------------------------------------------------------
#define LSTR 40

__device__ __forceinline__ void stage_tiles(
    unsigned short* __restrict__ dA, unsigned short* __restrict__ dB,
    const unsigned short* __restrict__ Xb, const unsigned short* __restrict__ Wb,
    int m0, int n0, int ks, int t) {
  const int row  = t >> 1;      // 0..127
  const int half = t & 1;       // each half = 16 bf16 = 32 B
  {
    const uint4* src = (const uint4*)(Xb + (size_t)(m0 + row) * HDIM + ks + half * 16);
    uint4 q0 = src[0], q1 = src[1];
    uint4* d = (uint4*)(dA + row * LSTR + half * 16);
    d[0] = q0; d[1] = q1;
  }
  {
    int n = n0 + row;
    uint4 q0 = make_uint4(0u, 0u, 0u, 0u), q1 = make_uint4(0u, 0u, 0u, 0u);
    if (n < VOCAB) {
      const uint4* src = (const uint4*)(Wb + (size_t)n * HDIM + ks + half * 16);
      q0 = src[0]; q1 = src[1];
    }
    uint4* d = (uint4*)(dB + row * LSTR + half * 16);
    d[0] = q0; d[1] = q1;
  }
}

// ISA 16-bit A-operand layout (16x32): lanes 0-15 hold K 0..7 & 16..23 of row
// M=lane; lanes 16-31 hold K 8..15 & 24..31 of row M=lane-16. Two contiguous
// 16-byte chunks per lane -> two ds_load_b128.
__device__ __forceinline__ v16bf load_frag(const unsigned short* __restrict__ base,
                                           int row, int lane) {
  const int c0 = (lane >> 4) << 3;      // element offset: 0 or 8
  union { uint4 q[2]; v16bf v; } u;
  u.q[0] = *(const uint4*)(base + row * LSTR + c0);
  u.q[1] = *(const uint4*)(base + row * LSTR + c0 + 16);
  return u.v;
}

__global__ __launch_bounds__(256) void gemm_wmma_bf16_kernel(
    const unsigned short* __restrict__ Xb,   // 2048 x 512 bf16
    const unsigned short* __restrict__ Wb,   // V    x 512 bf16
    const float* __restrict__ bias,
    float* __restrict__ out) {
  __shared__ unsigned short As[2][128 * LSTR];
  __shared__ unsigned short Bs[2][128 * LSTR];

  const int t    = threadIdx.x;
  const int lane = t & 31;
  const int wave = t >> 5;
  const int wm   = wave >> 2;               // 0..1 : M-group (64 rows)
  const int wn   = wave & 3;                // 0..3 : N-group (32 cols)
  const int m0   = blockIdx.y * 128;
  const int n0   = blockIdx.x * 128;

  v8f acc[4][2];
  #pragma unroll
  for (int i = 0; i < 4; ++i)
    #pragma unroll
    for (int j = 0; j < 2; ++j)
      #pragma unroll
      for (int e = 0; e < 8; ++e) acc[i][j][e] = 0.0f;

  stage_tiles(As[0], Bs[0], Xb, Wb, m0, n0, 0, t);
  __syncthreads();

  for (int ks = 0; ks < HDIM; ks += 32) {
    const int cur = (ks >> 5) & 1;
    if (ks + 32 < HDIM)
      stage_tiles(As[cur ^ 1], Bs[cur ^ 1], Xb, Wb, m0, n0, ks + 32, t);

    // hoist all fragment loads so WMMAs can issue back-to-back
    v16bf a[4], b[2];
    #pragma unroll
    for (int i = 0; i < 4; ++i)
      a[i] = load_frag(As[cur], wm * 64 + i * 16 + (lane & 15), lane);
    #pragma unroll
    for (int j = 0; j < 2; ++j)
      b[j] = load_frag(Bs[cur], wn * 32 + j * 16 + (lane & 15), lane);

    #pragma unroll
    for (int i = 0; i < 4; ++i)
      #pragma unroll
      for (int j = 0; j < 2; ++j)
        acc[i][j] = __builtin_amdgcn_wmma_f32_16x16x32_bf16(
            false, a[i], false, b[j], (short)0, acc[i][j], false, false);
    __syncthreads();
  }

  // C layout: VGPR e -> M = e (lanes 0-15) / e+8 (lanes 16-31), N = lane%16
  #pragma unroll
  for (int i = 0; i < 4; ++i) {
    const int rbase = m0 + wm * 64 + i * 16 + ((lane >> 4) << 3);
    #pragma unroll
    for (int j = 0; j < 2; ++j) {
      const int col = n0 + wn * 32 + j * 16 + (lane & 15);
      if (col < VOCAB) {
        const float bj = bias[col];
        #pragma unroll
        for (int e = 0; e < 8; ++e)
          out[(size_t)(rbase + e) * VOCAB + col] = (acc[i][j][e] + bj) * INVT;
      }
    }
  }
}

// ---------------------------------------------------------------------------
// Per-row top-k(50) / top-p(0.9) / softmax, in place on d_out.
// Row (201 KB) staged in LDS via the Tensor Data Mover; 3-level radix
// histogram select of the 50th value; bitonic sort of survivors; final
// streamed probability write.
// ---------------------------------------------------------------------------
__device__ __forceinline__ unsigned fmap(float f) {   // order-preserving map
  unsigned u = __float_as_uint(f);
  return (u & 0x80000000u) ? ~u : (u | 0x80000000u);
}
__device__ __forceinline__ float funmap(unsigned u) {
  unsigned b = (u & 0x80000000u) ? (u & 0x7FFFFFFFu) : ~u;
  return __uint_as_float(b);
}

#define ROWPAD 50272   // VOCAB rounded up (16B align)

__global__ __launch_bounds__(512) void topk_topp_softmax_kernel(float* __restrict__ logits) {
  extern __shared__ float smem[];
  float*    rowv = smem;                          // ROWPAD floats
  unsigned* hist = (unsigned*)(smem + ROWPAD);    // 2048
  unsigned* cand = hist + 2048;                   // 4096
  float*    srt  = (float*)(cand + 4096);         // 128

  __shared__ int      s_ncand, s_nsurv, s_b1, s_need2, s_b2, s_need3;
  __shared__ unsigned s_thresh;
  __shared__ float    s_m, s_vcut, s_invZ;

  const int    tid     = threadIdx.x;
  const size_t rowbase = (size_t)blockIdx.x * VOCAB;

  // ---- TDM DMA: global row -> LDS (issued by wave 0 only) ----
  if (tid < 32) {
    unsigned long long ga = (unsigned long long)(size_t)(logits + rowbase);
    unsigned lds_off = (unsigned)(size_t)rowv;    // low 32 bits = LDS byte offset
    // D# group0: count=1 | lds_addr | global_addr[56:0] | type=2
    u32x4 g0 = { 1u,
                 lds_off,
                 (unsigned)ga,
                 ((unsigned)((ga >> 32) & 0x01FFFFFFull)) | 0x80000000u };
    // D# group1: data_size=4B; tensor_dim0=V, tensor_dim1=1; tile=V x 1;
    // tensor_dim0_stride=V
    i32x8 g1 = { (int)(2u << 16),
                 (int)(((unsigned)VOCAB & 0xFFFFu) << 16),
                 (int)((((unsigned)VOCAB >> 16) & 0xFFFFu) | (1u << 16)),
                 (int)(((unsigned)VOCAB & 0xFFFFu) << 16),
                 1,
                 VOCAB,
                 0, 0 };
    i32x4 gz4 = { 0, 0, 0, 0 };
    i32x8 gz8 = { 0, 0, 0, 0, 0, 0, 0, 0 };
    __builtin_amdgcn_tensor_load_to_lds(g0, g1, gz4, gz4, gz8, 0);
    __builtin_amdgcn_s_wait_tensorcnt(0);
  }
  __syncthreads();

  // ---- level 1: 11-bit MSB histogram over monotonic uints ----
  for (int i = tid; i < 2048; i += 512) hist[i] = 0u;
  if (tid == 0) s_ncand = 0;
  __syncthreads();
  for (int i = tid; i < VOCAB; i += 512)
    atomicAdd(&hist[fmap(rowv[i]) >> 21], 1u);
  __syncthreads();
  if (tid == 0) {
    unsigned cum = 0; int b = 2047;
    for (; b >= 0; --b) {
      unsigned c = hist[b];
      if (cum + c >= (unsigned)TOPK) { s_b1 = b; s_need2 = TOPK - (int)cum; break; }
      cum += c;
    }
    if (b < 0) { s_b1 = 0; s_need2 = 1; }
  }
  __syncthreads();

  // ---- collect candidates: everything in/above the cutoff bin ----
  const int b1 = s_b1;
  for (int i = tid; i < VOCAB; i += 512) {
    unsigned u = fmap(rowv[i]);
    if ((int)(u >> 21) >= b1) {
      int idx = atomicAdd(&s_ncand, 1);
      if (idx < 4096) cand[idx] = u;
    }
  }
  __syncthreads();
  const int ncand = min(s_ncand, 4096);

  // ---- level 2: bits [20:10] within bin b1 ----
  for (int i = tid; i < 2048; i += 512) hist[i] = 0u;
  __syncthreads();
  for (int i = tid; i < ncand; i += 512) {
    unsigned u = cand[i];
    if ((int)(u >> 21) == b1) atomicAdd(&hist[(u >> 10) & 0x7FFu], 1u);
  }
  __syncthreads();
  if (tid == 0) {
    unsigned need = (unsigned)s_need2, cum = 0; int b = 2047;
    for (; b >= 0; --b) {
      unsigned c = hist[b];
      if (cum + c >= need) { s_b2 = b; s_need3 = (int)(need - cum); break; }
      cum += c;
    }
    if (b < 0) { s_b2 = 0; s_need3 = 1; }
  }
  __syncthreads();

  // ---- level 3: bits [9:0] -> exact 50th-largest value ----
  const int b2 = s_b2;
  const unsigned pref22 = ((unsigned)b1 << 11) | (unsigned)b2;
  for (int i = tid; i < 1024; i += 512) hist[i] = 0u;
  __syncthreads();
  for (int i = tid; i < ncand; i += 512) {
    unsigned u = cand[i];
    if ((u >> 10) == pref22) atomicAdd(&hist[u & 0x3FFu], 1u);
  }
  __syncthreads();
  if (tid == 0) {
    unsigned need = (unsigned)s_need3, cum = 0; int b = 1023;
    for (; b >= 0; --b) {
      unsigned c = hist[b];
      if (cum + c >= need) break;
      cum += c;
    }
    if (b < 0) b = 0;
    s_thresh = (pref22 << 10) | (unsigned)b;
    s_nsurv  = 0;
  }
  __syncthreads();

  // ---- survivors (>= v50) as floats ----
  const unsigned thr = s_thresh;
  for (int i = tid; i < ncand; i += 512) {
    unsigned u = cand[i];
    if (u >= thr) {
      int idx = atomicAdd(&s_nsurv, 1);
      if (idx < 128) srt[idx] = funmap(u);
    }
  }
  __syncthreads();
  const int ns = min(s_nsurv, 128);
  if (tid < 128 && tid >= ns) srt[tid] = -__builtin_inff();
  __syncthreads();

  // ---- bitonic sort, descending, 128 elements ----
  for (int k = 2; k <= 128; k <<= 1)
    for (int j = k >> 1; j > 0; j >>= 1) {
      if (tid < 128) {
        int ixj = tid ^ j;
        if (ixj > tid) {
          float a = srt[tid], b = srt[ixj];
          bool desc = ((tid & k) == 0);
          if (desc ? (a < b) : (a > b)) { srt[tid] = b; srt[ixj] = a; }
        }
      }
      __syncthreads();
    }

  // ---- top-p cutoff over the sorted top-k set ----
  if (tid == 0) {
    float m = srt[0];
    float Z = 0.f;
    for (int i = 0; i < ns; ++i) Z += __expf(srt[i] - m);
    float cum = 0.f; int nk = 0;
    for (int i = 0; i < ns; ++i) {
      cum += __expf(srt[i] - m) / Z;
      if (cum <= TOPP) nk = i + 1; else break;   // masked iff cum > P
    }
    float Z2 = 0.f;
    for (int i = 0; i < nk; ++i) Z2 += __expf(srt[i] - m);
    s_m    = m;
    s_vcut = (nk > 0) ? srt[nk - 1] : __builtin_inff();
    s_invZ = (nk > 0) ? (1.0f / Z2) : 0.0f;
  }
  __syncthreads();

  // ---- final streamed probability write ----
  const float m = s_m, vcut = s_vcut, invZ = s_invZ;
  for (int i = tid; i < VOCAB; i += 512) {
    float xv = rowv[i];
    logits[rowbase + i] = (xv >= vcut) ? __expf(xv - m) * invZ : 0.0f;
  }
}

// ---------------------------------------------------------------------------
// launch
// ---------------------------------------------------------------------------
extern "C" void kernel_launch(void* const* d_in, const int* in_sizes, int n_in,
                              void* d_out, int out_size, void* d_ws, size_t ws_size,
                              hipStream_t stream) {
  const float* x    = (const float*)d_in[0];   // 2048 x 512
  const float* W    = (const float*)d_in[1];   // V x 512
  const float* bias = (const float*)d_in[2];   // V
  float*       out  = (float*)d_out;           // 2048 x V

  unsigned short* Wbf = (unsigned short*)d_ws;                         // V*512 bf16
  unsigned short* Xbf = Wbf + (size_t)VOCAB * HDIM;                    // 2048*512 bf16

  const int nW4 = (VOCAB * HDIM) / 4;
  const int nX4 = (MROWS * HDIM) / 4;
  cvt_f32_to_bf16_kernel<<<(nW4 + 255) / 256, 256, 0, stream>>>(W, Wbf, nW4);
  cvt_f32_to_bf16_kernel<<<(nX4 + 255) / 256, 256, 0, stream>>>(x, Xbf, nX4);

  dim3 grid((VOCAB + 127) / 128, MROWS / 128);
  gemm_wmma_bf16_kernel<<<grid, 256, 0, stream>>>(Xbf, Wbf, bias, out);

  const size_t smem = (size_t)ROWPAD * 4 + 2048 * 4 + 4096 * 4 + 128 * 4;
  (void)hipFuncSetAttribute((const void*)topk_topp_softmax_kernel,
                            hipFuncAttributeMaxDynamicSharedMemorySize, (int)smem);
  topk_topp_softmax_kernel<<<MROWS, 512, smem, stream>>>(out);
}
